// MultiheadSelfAttention_65524021068000
// MI455X (gfx1250) — matrix operand order repaired and checked by reading, not compiled, not run on hardware
//
#include <hip/hip_runtime.h>

// ---------------- problem constants ----------------
#define B_SZ   2
#define T_SEQ  2048
#define M_DIM  2048
#define NHEAD  16
#define DHEAD  128
#define KHD    (NHEAD * DHEAD)   // 2048

typedef __bf16 bf16;
typedef __attribute__((ext_vector_type(16))) __bf16 v16bf;
typedef __attribute__((ext_vector_type(8)))  __bf16 v8bf;
typedef __attribute__((ext_vector_type(8)))  float   v8f;

union BF16x16 { v16bf v; v8bf h[2]; };

static __device__ __forceinline__ v8f v8f_zero() {
  v8f z;
#pragma unroll
  for (int i = 0; i < 8; ++i) z[i] = 0.0f;
  return z;
}

static __device__ __forceinline__ v8f wmma_bf16(v16bf a, v16bf b, v8f c) {
  // D = A(16x32 bf16) * B(32x16 bf16) + C(16x16 f32)
  return __builtin_amdgcn_wmma_f32_16x16x32_bf16(
      /*neg_a=*/false, a, /*neg_b=*/false, b,
      /*c_mod=*/(short)0, c, /*reuse_a=*/false, /*reuse_b=*/false);
}

// =====================================================================
// Prep kernels: fp32 -> bf16 conversions and layout transposes.
// Bandwidth-trivial (< 200 MB total at 23.3 TB/s).
// =====================================================================
__global__ void cvt_f32_to_bf16(const float* __restrict__ in,
                                bf16* __restrict__ out, int n) {
  int i = blockIdx.x * blockDim.x + threadIdx.x;
  if (i < n) out[i] = (bf16)in[i];
}

// [H][M][D] f32 -> [H][D][M] bf16   (B-matrix layout for projections)
__global__ void transpose_w_hmd(const float* __restrict__ in,
                                bf16* __restrict__ out, int n) {
  int i = blockIdx.x * blockDim.x + threadIdx.x;
  if (i >= n) return;
  int m = i % M_DIM;
  int t = i / M_DIM;
  int d = t % DHEAD;
  int h = t / DHEAD;
  out[i] = (bf16)in[((size_t)h * M_DIM + m) * DHEAD + d];
}

// w_ao [H*D][M] f32 -> [M][H*D] bf16  (B-matrix layout for output proj)
__global__ void transpose_wao(const float* __restrict__ in,
                              bf16* __restrict__ out, int n) {
  int i = blockIdx.x * blockDim.x + threadIdx.x;
  if (i >= n) return;
  int k = i % KHD;
  int m = i / KHD;
  out[i] = (bf16)in[(size_t)k * M_DIM + m];
}

// V [BH][T][D] bf16 -> Vt [BH][D][T] bf16  (B-matrix layout for P*V)
__global__ void transpose_v(const bf16* __restrict__ V,
                            bf16* __restrict__ Vt, int n) {
  int i = blockIdx.x * blockDim.x + threadIdx.x;
  if (i >= n) return;
  int t  = i % T_SEQ;
  int r  = i / T_SEQ;
  int d  = r % DHEAD;
  int bh = r / DHEAD;
  Vt[i] = V[((size_t)bh * T_SEQ + t) * DHEAD + d];
}

// RoPE (pairs d, d+64) applied in place to Q and K; folds the 1/128
// muP scale into Q. One thread per (bh, t, d<64).
__global__ void rope_kernel(bf16* __restrict__ Q, bf16* __restrict__ K, int n) {
  int i = blockIdx.x * blockDim.x + threadIdx.x;
  if (i >= n) return;
  int d  = i & 63;
  int t  = (i >> 6) % T_SEQ;
  float ang = (float)t * __powf(10000.0f, -(float)d * (1.0f / 64.0f));
  float s, c;
  __sincosf(ang, &s, &c);
  size_t base = ((size_t)(i >> 6)) * (size_t)DHEAD;  // (bh*T + t) * 128
  float qe = (float)Q[base + d], qo = (float)Q[base + d + 64];
  Q[base + d]      = (bf16)((qe * c - qo * s) * (1.0f / 128.0f));
  Q[base + d + 64] = (bf16)((qe * s + qo * c) * (1.0f / 128.0f));
  float ke = (float)K[base + d], ko = (float)K[base + d + 64];
  K[base + d]      = (bf16)(ke * c - ko * s);
  K[base + d + 64] = (bf16)(ke * s + ko * c);
}

// =====================================================================
// Generic 64x128 block-tile bf16 GEMM (8 waves, wave tile 32x32).
// A: row-major [rows][K] bf16. Bm: "[n][k]" layout bf16 (k contiguous).
// =====================================================================
template <bool BF16_OUT>
static __device__ __forceinline__ void gemm_tile_64x128(
    const bf16* __restrict__ A, int lda,
    const bf16* __restrict__ Bm, int ldb,
    void* __restrict__ Cv, int ldc, int Kdim, int row0, int col0) {
  const int lane = threadIdx.x & 31;
  const int wave = threadIdx.x >> 5;
  const int hi = lane >> 4;   // half-wave select
  const int ln = lane & 15;
  const int r0 = row0 + (wave & 1) * 32;   // 2 row groups of 32
  const int c0 = col0 + (wave >> 1) * 32;  // 4 col groups of 32

  v8f acc[2][2];
#pragma unroll
  for (int mt = 0; mt < 2; ++mt)
#pragma unroll
    for (int nt = 0; nt < 2; ++nt) acc[mt][nt] = v8f_zero();

  for (int k0 = 0; k0 < Kdim; k0 += 32) {
    v16bf a[2], b[2];
#pragma unroll
    for (int mt = 0; mt < 2; ++mt) {
      const bf16* ar = A + (size_t)(r0 + mt * 16 + ln) * lda + k0 + hi * 16;
      BF16x16 t;
      t.h[0] = *(const v8bf*)(ar);
      t.h[1] = *(const v8bf*)(ar + 8);
      a[mt] = t.v;
    }
#pragma unroll
    for (int nt = 0; nt < 2; ++nt) {
      const bf16* br = Bm + (size_t)(c0 + nt * 16 + ln) * ldb + k0 + hi * 16;
      b[nt] = *(const v16bf*)br;
    }
#pragma unroll
    for (int mt = 0; mt < 2; ++mt)
#pragma unroll
      for (int nt = 0; nt < 2; ++nt)
        acc[mt][nt] = wmma_bf16(a[mt], b[nt], acc[mt][nt]);
  }

#pragma unroll
  for (int mt = 0; mt < 2; ++mt)
#pragma unroll
    for (int nt = 0; nt < 2; ++nt)
#pragma unroll
      for (int v = 0; v < 8; ++v) {
        int r = r0 + mt * 16 + v + 8 * hi;  // C layout: VGPR v -> row v(+8)
        int c = c0 + nt * 16 + ln;          // lane&15 -> column
        if (BF16_OUT)
          ((bf16*)Cv)[(size_t)r * ldc + c] = (bf16)acc[mt][nt][v];
        else
          ((float*)Cv)[(size_t)r * ldc + c] = acc[mt][nt][v];
      }
}

// q/k/v projection: z encodes (b, h, which-of-qkv)
__global__ __launch_bounds__(256) void qkv_proj_kernel(
    const bf16* __restrict__ xb, const bf16* __restrict__ wqT,
    const bf16* __restrict__ wkT, const bf16* __restrict__ wvT,
    bf16* __restrict__ Q, bf16* __restrict__ K, bf16* __restrict__ V) {
  int z = blockIdx.z;
  int s = z % 3;
  int bh = z / 3;
  int b = bh / NHEAD, h = bh % NHEAD;
  const bf16* W = (s == 0 ? wqT : s == 1 ? wkT : wvT) + (size_t)h * DHEAD * M_DIM;
  bf16* C = (s == 0 ? Q : s == 1 ? K : V) + (size_t)bh * T_SEQ * DHEAD;
  const bf16* A = xb + (size_t)b * T_SEQ * M_DIM;
  gemm_tile_64x128<true>(A, M_DIM, W, M_DIM, C, DHEAD, M_DIM,
                         blockIdx.x * 64, 0);
}

// output projection: r[B*T][M] = O[B*T][HD] x waoT
__global__ __launch_bounds__(256) void out_proj_kernel(
    const bf16* __restrict__ O, const bf16* __restrict__ waoT,
    float* __restrict__ R) {
  gemm_tile_64x128<false>(O, KHD, waoT, KHD, R, M_DIM, KHD,
                          blockIdx.x * 64, blockIdx.y * 128);
}

// =====================================================================
// Causal flash attention. Block = 8 waves; each wave owns 16 query rows
// and the full 16x128 fp32 O accumulator. j advances in tiles of 32.
// =====================================================================
__global__ __launch_bounds__(256) void attn_kernel(
    const bf16* __restrict__ Qg, const bf16* __restrict__ Kg,
    const bf16* __restrict__ Vtg, bf16* __restrict__ Og) {
  __shared__ __align__(32) bf16 ldsP[8][16][32];  // per-wave P staging (8 KB)

  const int lane = threadIdx.x & 31;
  const int wave = threadIdx.x >> 5;
  const int hi = lane >> 4;
  const int ln = lane & 15;
  const int bh = blockIdx.y;                      // b*NHEAD + h
  const int i0 = blockIdx.x * 128 + wave * 16;    // query row base of wave

  const bf16* q  = Qg  + (size_t)bh * T_SEQ * DHEAD;
  const bf16* k  = Kg  + (size_t)bh * T_SEQ * DHEAD;
  const bf16* vt = Vtg + (size_t)bh * DHEAD * T_SEQ;

  // Preload Q A-fragments for all 4 k-chunks of D=128 (fixed for wave).
  v16bf qa[4];
  {
    const bf16* qrow = q + (size_t)(i0 + ln) * DHEAD + hi * 16;
#pragma unroll
    for (int kk = 0; kk < 4; ++kk) {
      BF16x16 t;
      t.h[0] = *(const v8bf*)(qrow + kk * 32);
      t.h[1] = *(const v8bf*)(qrow + kk * 32 + 8);
      qa[kk] = t.v;
    }
  }

  v8f o_acc[8];
#pragma unroll
  for (int i = 0; i < 8; ++i) o_acc[i] = v8f_zero();
  float rowM[8], rowL[8];
#pragma unroll
  for (int v = 0; v < 8; ++v) { rowM[v] = -3.0e38f; rowL[v] = 0.0f; }

  const int jmax = i0 + 15;
  for (int j0 = 0; j0 <= jmax; j0 += 32) {
    // ---- S = Q * K^T : two 16x16 tiles covering j0..j0+31 ----
    v8f s0 = v8f_zero(), s1 = v8f_zero();
#pragma unroll
    for (int kk = 0; kk < 4; ++kk) {
      v16bf b0 = *(const v16bf*)(k + (size_t)(j0 + ln) * DHEAD + kk * 32 + hi * 16);
      v16bf b1 = *(const v16bf*)(k + (size_t)(j0 + 16 + ln) * DHEAD + kk * 32 + hi * 16);
      s0 = wmma_bf16(qa[kk], b0, s0);
      s1 = wmma_bf16(qa[kk], b1, s1);
    }

    // ---- causal mask (only tiles touching diagonal) ----
    if (j0 + 31 > i0) {
#pragma unroll
      for (int v = 0; v < 8; ++v) {
        int irow = i0 + v + 8 * hi;
        if (j0 + ln > irow)      s0[v] = -1.0e30f;
        if (j0 + 16 + ln > irow) s1[v] = -1.0e30f;
      }
    }

    // ---- online softmax (rows live in half-wave groups of 16 lanes) ----
#pragma unroll
    for (int v = 0; v < 8; ++v) {
      float mx = fmaxf(s0[v], s1[v]);
#pragma unroll
      for (int off = 1; off < 16; off <<= 1)
        mx = fmaxf(mx, __shfl_xor(mx, off, 32));
      float mnew = fmaxf(rowM[v], mx);
      float sc = __expf(rowM[v] - mnew);
      rowM[v] = mnew;
      float p0 = __expf(s0[v] - mnew);
      float p1 = __expf(s1[v] - mnew);
      float rs = p0 + p1;
#pragma unroll
      for (int off = 1; off < 16; off <<= 1) rs += __shfl_xor(rs, off, 32);
      rowL[v] = rowL[v] * sc + rs;
      s0[v] = p0;
      s1[v] = p1;
#pragma unroll
      for (int f = 0; f < 8; ++f) o_acc[f][v] *= sc;
    }

    // ---- P: C-layout -> A-layout via private LDS tile ----
#pragma unroll
    for (int v = 0; v < 8; ++v) {
      int m = v + 8 * hi;
      ldsP[wave][m][ln]      = (bf16)s0[v];
      ldsP[wave][m][16 + ln] = (bf16)s1[v];
    }
    asm volatile("s_wait_dscnt 0" ::: "memory");  // cross-lane LDS hazard
    v16bf pa;
    {
      BF16x16 t;
      const bf16* pr = &ldsP[wave][ln][hi * 16];
      t.h[0] = *(const v8bf*)(pr);
      t.h[1] = *(const v8bf*)(pr + 8);
      pa = t.v;
    }
    asm volatile("s_wait_dscnt 0" ::: "memory");

    // ---- O += P * V  (Vt gives contiguous B-fragments) ----
#pragma unroll
    for (int dt = 0; dt < 8; ++dt) {
      v16bf bv = *(const v16bf*)(vt + (size_t)(dt * 16 + ln) * T_SEQ + j0 + hi * 16);
      o_acc[dt] = wmma_bf16(pa, bv, o_acc[dt]);
    }
  }

  // ---- normalize and store O as [B][T][H][D] bf16 ----
  const int b = bh / NHEAD, h = bh % NHEAD;
  float inv[8];
#pragma unroll
  for (int v = 0; v < 8; ++v) inv[v] = 1.0f / rowL[v];
#pragma unroll
  for (int dt = 0; dt < 8; ++dt)
#pragma unroll
    for (int v = 0; v < 8; ++v) {
      int m = v + 8 * hi;
      int d = dt * 16 + ln;
      Og[(((size_t)b * T_SEQ + i0 + m) * NHEAD + h) * DHEAD + d] =
          (bf16)(o_acc[dt][v] * inv[v]);
    }
}

// =====================================================================
// Host launch
// =====================================================================
extern "C" void kernel_launch(void* const* d_in, const int* in_sizes, int n_in,
                              void* d_out, int out_size, void* d_ws, size_t ws_size,
                              hipStream_t stream) {
  const float* x    = (const float*)d_in[0];
  const float* w_aq = (const float*)d_in[1];
  const float* w_ak = (const float*)d_in[2];
  const float* w_av = (const float*)d_in[3];
  const float* w_ao = (const float*)d_in[4];
  float* out = (float*)d_out;

  const size_t nX  = (size_t)B_SZ * T_SEQ * M_DIM;   // 8388608
  const size_t nW  = (size_t)NHEAD * M_DIM * DHEAD;  // 4194304
  const size_t nAO = (size_t)M_DIM * KHD;            // 4194304
  const size_t nQ  = (size_t)B_SZ * NHEAD * T_SEQ * DHEAD;  // 8388608

  char* p = (char*)d_ws;
  bf16* xb   = (bf16*)p; p += nX  * sizeof(bf16);
  bf16* wqT  = (bf16*)p; p += nW  * sizeof(bf16);
  bf16* wkT  = (bf16*)p; p += nW  * sizeof(bf16);
  bf16* wvT  = (bf16*)p; p += nW  * sizeof(bf16);
  bf16* waoT = (bf16*)p; p += nAO * sizeof(bf16);
  bf16* Q    = (bf16*)p; p += nQ  * sizeof(bf16);
  bf16* K    = (bf16*)p; p += nQ  * sizeof(bf16);
  bf16* V    = (bf16*)p; p += nQ  * sizeof(bf16);
  bf16* Vt   = (bf16*)p; p += nQ  * sizeof(bf16);
  bf16* O    = (bf16*)p; p += nQ  * sizeof(bf16);

  const int TPB = 256;
  cvt_f32_to_bf16<<<(int)((nX + TPB - 1) / TPB), TPB, 0, stream>>>(x, xb, (int)nX);
  transpose_w_hmd<<<(int)((nW + TPB - 1) / TPB), TPB, 0, stream>>>(w_aq, wqT, (int)nW);
  transpose_w_hmd<<<(int)((nW + TPB - 1) / TPB), TPB, 0, stream>>>(w_ak, wkT, (int)nW);
  transpose_w_hmd<<<(int)((nW + TPB - 1) / TPB), TPB, 0, stream>>>(w_av, wvT, (int)nW);
  transpose_wao<<<(int)((nAO + TPB - 1) / TPB), TPB, 0, stream>>>(w_ao, waoT, (int)nAO);

  // q/k/v projections: 64-row tiles x (B*H*3) targets
  dim3 gq(T_SEQ / 64, 1, B_SZ * NHEAD * 3);
  qkv_proj_kernel<<<gq, TPB, 0, stream>>>(xb, wqT, wkT, wvT, Q, K, V);

  // RoPE (+ 1/128 into Q)
  const size_t nR = (size_t)B_SZ * NHEAD * T_SEQ * 64;
  rope_kernel<<<(int)((nR + TPB - 1) / TPB), TPB, 0, stream>>>(Q, K, (int)nR);

  // V -> Vt
  transpose_v<<<(int)((nQ + TPB - 1) / TPB), TPB, 0, stream>>>(V, Vt, (int)nQ);

  // flash attention
  dim3 ga(T_SEQ / 128, B_SZ * NHEAD);
  attn_kernel<<<ga, TPB, 0, stream>>>(Q, K, Vt, O);

  // output projection -> fp32 result
  dim3 go((B_SZ * T_SEQ) / 64, M_DIM / 128);
  out_proj_kernel<<<go, TPB, 0, stream>>>(O, waoT, out);
}
